// BaseCTGANGenerator_3547642986885
// MI455X (gfx1250) — matrix operand order repaired and checked by reading, not compiled, not run on hardware
//
#include <hip/hip_runtime.h>
#include <hip/hip_bf16.h>
#include <stdint.h>

// ---------------- CDNA5 WMMA types ----------------
typedef __attribute__((ext_vector_type(16))) __bf16 v16bf;
typedef __attribute__((ext_vector_type(8)))  float  v8f;

constexpr int Bn   = 32768;   // batch
constexpr int Zn   = 128;     // latent
constexpr int Hn   = 256;     // hidden
constexpr int K2n  = 384;     // Z + H
constexpr int K3n  = 640;     // Z + 2H
constexpr int NOn  = 1550;    // output features
constexpr int NOPn = 1568;    // padded to 98 * 16
constexpr int APAD = 8;       // LDS A-tile row pad (ushorts) to break bank conflicts
constexpr float TAU_INV = 5.0f;   // 1 / 0.2
constexpr float BN_EPS  = 1e-3f;

// ---------------- helpers ----------------
__device__ __forceinline__ unsigned short f2bf(float f) {
  union { float f; uint32_t u; } c; c.f = f;
  uint32_t u = c.u;
  uint32_t r = (u + 0x7FFFu + ((u >> 16) & 1u)) >> 16;   // RNE
  return (unsigned short)r;
}

__device__ __forceinline__ uint32_t mix32(uint32_t x) {
  x ^= x >> 17; x *= 0xED5AD4BBu;
  x ^= x >> 11; x *= 0xAC4C1B51u;
  x ^= x >> 15; x *= 0x31848BABu;
  x ^= x >> 14; return x;
}

// A fragment: 16x32 bf16 tile from LDS (row-major, ld ushorts per row).
// ISA 7.12.2 16-bit A layout: lane half h, row r=lane&15;
//   VGPR i<4 : K = h*8 + 2i, 2i+1 ; VGPR i>=4 : K = 16 + h*8 + 2(i-4) ...
__device__ __forceinline__ v16bf a_frag(const unsigned short* base, int ld, int lane) {
  const int half = (lane >> 4) & 1, r = lane & 15;
  const unsigned short* p = base + r * ld;
  union { uint32_t u[8]; v16bf v; } cv;
#pragma unroll
  for (int i = 0; i < 8; ++i) {
    const int kb = ((i >> 2) << 4) + half * 8 + ((i & 3) << 1);
    cv.u[i] = *(const uint32_t*)(p + kb);
  }
  return cv.v;
}

// B fragment: packed weights, 512 bf16 per (ntile,kstep), 16 contiguous per lane.
__device__ __forceinline__ v16bf b_frag(const unsigned short* wp, int lane) {
  const uint4* q = (const uint4*)(wp + lane * 16);
  union { uint4 u[2]; v16bf v; } cv;
  cv.u[0] = q[0]; cv.u[1] = q[1];
  return cv.v;
}

__device__ __forceinline__ v8f wmma_bf16(v16bf a, v16bf b, v8f c) {
  return __builtin_amdgcn_wmma_f32_16x16x32_bf16(false, a, false, b, (short)0, c, false, false);
}

// ---------------- prep kernels ----------------
__global__ void k_zero(float* __restrict__ p, int n) {
  int i = blockIdx.x * blockDim.x + threadIdx.x;
  if (i < n) p[i] = 0.f;
}

// Pack row-major W[K][N] (f32) into B-fragment-major bf16:
// out[((ntile*nk + ks)*32 + lane)*16 + j] = W[ks*32 + (j>>3)*16 + (lane>>4)*8 + (j&7)][ntile*16 + (lane&15)]
__global__ void k_pack(const float* __restrict__ W, unsigned short* __restrict__ out,
                       int K, int N, int Npad) {
  const int nk = K >> 5, nt = Npad >> 4;
  const long total = (long)nt * nk * 512;
  for (long idx = (long)blockIdx.x * blockDim.x + threadIdx.x; idx < total;
       idx += (long)gridDim.x * blockDim.x) {
    const int j    = (int)(idx & 15);
    const int lane = (int)((idx >> 4) & 31);
    const long t   = idx >> 9;
    const int ks   = (int)(t % nk);
    const int nti  = (int)(t / nk);
    const int k = ks * 32 + (j >> 3) * 16 + ((lane >> 4) & 1) * 8 + (j & 7);
    const int n = nti * 16 + (lane & 15);
    const float v = (n < N) ? W[(long)k * N + n] : 0.0f;
    out[idx] = f2bf(v);
  }
}

__global__ void k_cvt_bf16(const float* __restrict__ in, unsigned short* __restrict__ out, long n) {
  for (long i = (long)blockIdx.x * blockDim.x + threadIdx.x; i < n;
       i += (long)gridDim.x * blockDim.x)
    out[i] = f2bf(in[i]);
}

// ---------------- BN stats ----------------
__global__ void k_colstats(const float* __restrict__ h, float* __restrict__ sum,
                           float* __restrict__ sq, int rows, int rpb) {
  const int c  = threadIdx.x;             // 256 columns = blockDim
  const int r0 = blockIdx.x * rpb;
  const int r1 = (r0 + rpb < rows) ? r0 + rpb : rows;
  float s = 0.f, q = 0.f;
  for (int r = r0; r < r1; ++r) {
    float v = h[(long)r * Hn + c];
    s += v; q += v * v;
  }
  atomicAdd(&sum[c], s);
  atomicAdd(&sq[c], q);
}

__global__ void k_finalize(const float* __restrict__ sum, const float* __restrict__ sq,
                           const float* __restrict__ g, const float* __restrict__ be,
                           float* __restrict__ scale, float* __restrict__ shift, int rows) {
  const int c = threadIdx.x;
  const float inv = 1.0f / (float)rows;
  const float mu  = sum[c] * inv;
  const float var = sq[c] * inv - mu * mu;
  const float rs  = rsqrtf(var + BN_EPS);
  const float sc  = g[c] * rs;
  scale[c] = sc;
  shift[c] = be[c] - mu * sc;
}

// ---------------- GEMM 1: h1 = z @ W1 + b1  (K=128, N=256, 64 rows/block) ----------------
__global__ __launch_bounds__(256) void k_gemm1(const unsigned short* __restrict__ zbf,
                                               const unsigned short* __restrict__ w1p,
                                               const float* __restrict__ b1,
                                               float* __restrict__ h1) {
  extern __shared__ __align__(16) char smem[];
  unsigned short* sA = (unsigned short*)smem;            // 64 x (128+APAD)
  const int ld = Zn + APAD;
  const int m0 = blockIdx.x * 64;
  for (int idx = threadIdx.x; idx < 64 * Zn; idx += blockDim.x) {
    int row = idx >> 7, c = idx & (Zn - 1);
    sA[row * ld + c] = zbf[(long)(m0 + row) * Zn + c];
  }
  __syncthreads();
  const int wave = threadIdx.x >> 5, lane = threadIdx.x & 31;
  const int half = lane >> 4, r = lane & 15;
  v8f acc[4][2];
#pragma unroll
  for (int mt = 0; mt < 4; ++mt)
#pragma unroll
    for (int t = 0; t < 2; ++t)
#pragma unroll
      for (int i = 0; i < 8; ++i) acc[mt][t][i] = 0.f;

  for (int ks = 0; ks < 4; ++ks) {
    v16bf bfr[2];
#pragma unroll
    for (int t = 0; t < 2; ++t)
      bfr[t] = b_frag(w1p + (long)((wave * 2 + t) * 4 + ks) * 512, lane);
#pragma unroll
    for (int mt = 0; mt < 4; ++mt) {
      v16bf afr = a_frag(sA + (mt * 16) * ld + ks * 32, ld, lane);
#pragma unroll
      for (int t = 0; t < 2; ++t) acc[mt][t] = wmma_bf16(afr, bfr[t], acc[mt][t]);
    }
  }
#pragma unroll
  for (int t = 0; t < 2; ++t) {
    const int col = (wave * 2 + t) * 16 + r;
    const float bias = b1[col];
#pragma unroll
    for (int mt = 0; mt < 4; ++mt)
#pragma unroll
      for (int i = 0; i < 8; ++i) {
        const int row = m0 + mt * 16 + half * 8 + i;
        h1[(long)row * Hn + col] = acc[mt][t][i] + bias;
      }
  }
}

// ---------------- GEMM 2: h2 = [bn1(h1)|z] @ W2 + b2  (K=384, N=256) ----------------
__global__ __launch_bounds__(256) void k_gemm2(const unsigned short* __restrict__ zbf,
                                               const float* __restrict__ h1,
                                               const float* __restrict__ scale1,
                                               const float* __restrict__ shift1,
                                               const unsigned short* __restrict__ w2p,
                                               const float* __restrict__ b2,
                                               float* __restrict__ h2) {
  extern __shared__ __align__(16) char smem[];
  unsigned short* sA = (unsigned short*)smem;            // 64 x (384+APAD)
  const int ld = K2n + APAD;
  const int m0 = blockIdx.x * 64;
  for (int idx = threadIdx.x; idx < 64 * Hn; idx += blockDim.x) {
    int row = idx >> 8, c = idx & (Hn - 1);
    float v = h1[(long)(m0 + row) * Hn + c];
    v = fmaxf(0.f, v * scale1[c] + shift1[c]);
    sA[row * ld + c] = f2bf(v);
  }
  for (int idx = threadIdx.x; idx < 64 * Zn; idx += blockDim.x) {
    int row = idx >> 7, c = idx & (Zn - 1);
    sA[row * ld + Hn + c] = zbf[(long)(m0 + row) * Zn + c];
  }
  __syncthreads();
  const int wave = threadIdx.x >> 5, lane = threadIdx.x & 31;
  const int half = lane >> 4, r = lane & 15;
  v8f acc[4][2];
#pragma unroll
  for (int mt = 0; mt < 4; ++mt)
#pragma unroll
    for (int t = 0; t < 2; ++t)
#pragma unroll
      for (int i = 0; i < 8; ++i) acc[mt][t][i] = 0.f;

  for (int ks = 0; ks < 12; ++ks) {
    v16bf bfr[2];
#pragma unroll
    for (int t = 0; t < 2; ++t)
      bfr[t] = b_frag(w2p + (long)((wave * 2 + t) * 12 + ks) * 512, lane);
#pragma unroll
    for (int mt = 0; mt < 4; ++mt) {
      v16bf afr = a_frag(sA + (mt * 16) * ld + ks * 32, ld, lane);
#pragma unroll
      for (int t = 0; t < 2; ++t) acc[mt][t] = wmma_bf16(afr, bfr[t], acc[mt][t]);
    }
  }
#pragma unroll
  for (int t = 0; t < 2; ++t) {
    const int col = (wave * 2 + t) * 16 + r;
    const float bias = b2[col];
#pragma unroll
    for (int mt = 0; mt < 4; ++mt)
#pragma unroll
      for (int i = 0; i < 8; ++i) {
        const int row = m0 + mt * 16 + half * 8 + i;
        h2[(long)row * Hn + col] = acc[mt][t][i] + bias;
      }
  }
}

// ---------------- activations ----------------
template <int LEN>
__device__ __forceinline__ void gsoftmax(const float* __restrict__ sC,
                                         const float* __restrict__ bo,
                                         float* __restrict__ out,
                                         int grow, int lrow, int c0, uint32_t salt) {
  float v[LEN];
  float mx = -3.402823e38f;
#pragma unroll
  for (int j = 0; j < LEN; ++j) {
    float x = sC[lrow * NOPn + c0 + j] + bo[c0 + j];
    uint32_t h = mix32(((uint32_t)(grow * NOn + c0 + j)) * 0x9E3779B1u ^ salt);
    float u = fmaxf((h >> 8) * (1.0f / 16777216.0f), 1e-7f);
    float t = -logf(u);                  // > 0
    float g = -logf(t);                  // gumbel(0,1)
    x = (x + g) * TAU_INV;
    v[j] = x;
    mx = fmaxf(mx, x);
  }
  float s = 0.f;
#pragma unroll
  for (int j = 0; j < LEN; ++j) { float e = expf(v[j] - mx); v[j] = e; s += e; }
  const float inv = 1.0f / s;
#pragma unroll
  for (int j = 0; j < LEN; ++j) out[(long)grow * NOn + c0 + j] = v[j] * inv;
}

// ---------------- GEMM 3 + head, fused (K=640, N=1550->1568, 32 rows/block) ----------------
__global__ __launch_bounds__(256) void k_gemm3_act(const unsigned short* __restrict__ zbf,
                                                   const float* __restrict__ h1,
                                                   const float* __restrict__ h2,
                                                   const float* __restrict__ scale1,
                                                   const float* __restrict__ shift1,
                                                   const float* __restrict__ scale2,
                                                   const float* __restrict__ shift2,
                                                   const unsigned short* __restrict__ wop,
                                                   const float* __restrict__ bo,
                                                   float* __restrict__ out) {
  extern __shared__ __align__(16) char smem[];
  const int ld = K3n + APAD;
  unsigned short* sA = (unsigned short*)smem;                       // 32 x (640+APAD)
  float* sC = (float*)(smem + (size_t)32 * ld * sizeof(unsigned short)); // 32 x 1568
  const int m0 = blockIdx.x * 32;

  // Build A = [ bn2relu(h2) (256) | bn1relu(h1) (256) | z (128) ] as bf16
  for (int idx = threadIdx.x; idx < 32 * Hn; idx += blockDim.x) {
    int row = idx >> 8, c = idx & (Hn - 1);
    float v2 = h2[(long)(m0 + row) * Hn + c];
    v2 = fmaxf(0.f, v2 * scale2[c] + shift2[c]);
    sA[row * ld + c] = f2bf(v2);
    float v1 = h1[(long)(m0 + row) * Hn + c];
    v1 = fmaxf(0.f, v1 * scale1[c] + shift1[c]);
    sA[row * ld + Hn + c] = f2bf(v1);
  }
  for (int idx = threadIdx.x; idx < 32 * Zn; idx += blockDim.x) {
    int row = idx >> 7, c = idx & (Zn - 1);
    sA[row * ld + 2 * Hn + c] = zbf[(long)(m0 + row) * Zn + c];
  }
  __syncthreads();

  const int wave = threadIdx.x >> 5, lane = threadIdx.x & 31;
  const int half = lane >> 4, r = lane & 15;

  for (int nt = wave; nt < NOPn / 16; nt += 8) {
    v8f acc0, acc1;
#pragma unroll
    for (int i = 0; i < 8; ++i) { acc0[i] = 0.f; acc1[i] = 0.f; }
    for (int ks = 0; ks < 20; ++ks) {
      v16bf bfr = b_frag(wop + (long)(nt * 20 + ks) * 512, lane);
      v16bf a0 = a_frag(sA + ks * 32, ld, lane);
      v16bf a1 = a_frag(sA + 16 * ld + ks * 32, ld, lane);
      acc0 = wmma_bf16(a0, bfr, acc0);
      acc1 = wmma_bf16(a1, bfr, acc1);
    }
#pragma unroll
    for (int i = 0; i < 8; ++i) {
      sC[(half * 8 + i) * NOPn + nt * 16 + r]        = acc0[i];
      sC[(16 + half * 8 + i) * NOPn + nt * 16 + r]   = acc1[i];
    }
  }
  __syncthreads();

  // tanh on alpha logits: 50 features, col = f*11
  for (int t = threadIdx.x; t < 32 * 50; t += blockDim.x) {
    int row = t / 50, f = t - row * 50, col = f * 11;
    float v = sC[row * NOPn + col] + bo[col];
    out[(long)(m0 + row) * NOn + col] = tanhf(v);
  }
  // gumbel-softmax over beta logits (groups of 10)
  for (int t = threadIdx.x; t < 32 * 50; t += blockDim.x) {
    int row = t / 50, f = t - row * 50;
    gsoftmax<10>(sC, bo, out, m0 + row, row, f * 11 + 1, 0x85EBCA6Bu);
  }
  // gumbel-softmax over categorical logits (groups of 20)
  for (int t = threadIdx.x; t < 32 * 50; t += blockDim.x) {
    int row = t / 50, f = t - row * 50;
    gsoftmax<20>(sC, bo, out, m0 + row, row, 550 + f * 20, 0xC2B2AE35u);
  }
}

// ---------------- launch ----------------
extern "C" void kernel_launch(void* const* d_in, const int* in_sizes, int n_in,
                              void* d_out, int out_size, void* d_ws, size_t ws_size,
                              hipStream_t stream) {
  const float* z   = (const float*)d_in[0];
  const float* W1  = (const float*)d_in[1];
  const float* b1  = (const float*)d_in[2];
  const float* g1  = (const float*)d_in[3];
  const float* be1 = (const float*)d_in[4];
  const float* W2  = (const float*)d_in[5];
  const float* b2  = (const float*)d_in[6];
  const float* g2  = (const float*)d_in[7];
  const float* be2 = (const float*)d_in[8];
  const float* Wo  = (const float*)d_in[9];
  const float* bo  = (const float*)d_in[10];
  float* out = (float*)d_out;
  (void)in_sizes; (void)n_in; (void)out_size; (void)ws_size;

  char* ws = (char*)d_ws;
  size_t off = 0;
  unsigned short* zbf = (unsigned short*)(ws + off); off += (size_t)Bn * Zn * 2;
  float* h1 = (float*)(ws + off); off += (size_t)Bn * Hn * 4;
  float* h2 = (float*)(ws + off); off += (size_t)Bn * Hn * 4;
  float* st = (float*)(ws + off); off += 8 * 256 * 4;
  unsigned short* w1p = (unsigned short*)(ws + off); off += (size_t)Zn * Hn * 2;
  unsigned short* w2p = (unsigned short*)(ws + off); off += (size_t)K2n * Hn * 2;
  unsigned short* wop = (unsigned short*)(ws + off); off += (size_t)K3n * NOPn * 2;

  float* s1sum = st + 0,    *s1sq = st + 256;
  float* s2sum = st + 512,  *s2sq = st + 768;
  float* sc1   = st + 1024, *sh1  = st + 1280;
  float* sc2   = st + 1536, *sh2  = st + 1792;

  // phase 0: prep (zero stats, pack weights to fragment-major bf16, z -> bf16)
  k_zero<<<4, 256, 0, stream>>>(st, 1024);
  k_pack<<<64, 256, 0, stream>>>(W1, w1p, Zn, Hn, Hn);
  k_pack<<<192, 256, 0, stream>>>(W2, w2p, K2n, Hn, Hn);
  k_pack<<<1024, 256, 0, stream>>>(Wo, wop, K3n, NOn, NOPn);
  k_cvt_bf16<<<2048, 256, 0, stream>>>(z, zbf, (long)Bn * Zn);

  // phase 1
  k_gemm1<<<Bn / 64, 256, (size_t)64 * (Zn + APAD) * 2, stream>>>(zbf, w1p, b1, h1);
  k_colstats<<<32, 256, 0, stream>>>(h1, s1sum, s1sq, Bn, 1024);
  k_finalize<<<1, 256, 0, stream>>>(s1sum, s1sq, g1, be1, sc1, sh1, Bn);

  // phase 2
  k_gemm2<<<Bn / 64, 256, (size_t)64 * (K2n + APAD) * 2, stream>>>(zbf, h1, sc1, sh1, w2p, b2, h2);
  k_colstats<<<32, 256, 0, stream>>>(h2, s2sum, s2sq, Bn, 1024);
  k_finalize<<<1, 256, 0, stream>>>(s2sum, s2sq, g2, be2, sc2, sh2, Bn);

  // phase 3: fused big GEMM + head (needs ~237 KB LDS -> one WG per WGP)
  const int smem3 = 32 * (K3n + APAD) * 2 + 32 * NOPn * 4;
  hipFuncSetAttribute((const void*)k_gemm3_act,
                      hipFuncAttributeMaxDynamicSharedMemorySize, smem3);
  k_gemm3_act<<<Bn / 32, 256, (size_t)smem3, stream>>>(zbf, h1, h2, sc1, sh1, sc2, sh2,
                                                       wop, bo, out);
}
